// QuantizationLayer_32100585570575
// MI455X (gfx1250) — compile-verified, matrix operand order checked by reading, not run on hardware
//
#include <hip/hip_runtime.h>
#include <hip/hip_bf16.h>
#include <math.h>

typedef _Float16 v8h  __attribute__((ext_vector_type(8)));
typedef _Float16 v16h __attribute__((ext_vector_type(16)));
typedef float    v8f  __attribute__((ext_vector_type(8)));

constexpr int kW = 320, kH = 240, kHW = 76800;
constexpr int kS = 48, kSegLen = 8192, kNev = 393216, kB = 32;
constexpr int kSIDX = 4, kEIDX = 38, kNScan = 33;   // EIDX - (SIDX+1) = 33 scan rows
constexpr int kDpad = 320;                          // hist rows padded to 320 (Y zero-padded)
constexpr int kMaskW = kHW / 32;                    // 2400 dwords per bitmask
constexpr int kWtPitch = 328;                       // LDS row pitch (halves) for bank spread

// ---------------------------------------------------------------------------
// Kernel 0: zero the container accumulators
// ---------------------------------------------------------------------------
__global__ void k_zero(unsigned* __restrict__ p, size_t n) {
  size_t i = (size_t)blockIdx.x * 256 + threadIdx.x;
  if (i < n) p[i] = 0u;
}

// ---------------------------------------------------------------------------
// Kernel 1: per-(batch,segment) histograms of x and y via LDS atomics,
//           pack (x,y) into one u32 per event for the scan phase,
//           emit histograms as f16 (counts < 2048 -> exact) for WMMA.
// ---------------------------------------------------------------------------
__global__ __launch_bounds__(256)
void k_hist(const int* __restrict__ ev, unsigned* __restrict__ packed,
            _Float16* __restrict__ hx, _Float16* __restrict__ hy) {
  __shared__ unsigned shx[kDpad], shy[kDpad];
  int bs = blockIdx.x;               // b * 48 + s
  int t = threadIdx.x;
  for (int d = t; d < kDpad; d += 256) { shx[d] = 0u; shy[d] = 0u; }
  __syncthreads();
  int b = bs / kS, s = bs % kS;
  size_t base = (size_t)b * kNev + (size_t)s * kSegLen;
  for (int i = t; i < kSegLen; i += 256) {
    size_t e5 = (base + i) * 5;
    int x = ev[e5];                  // [0, 320)
    int y = ev[e5 + 1];              // [0, 240)
    atomicAdd(&shx[x], 1u);
    atomicAdd(&shy[y], 1u);
    packed[base + i] = (unsigned)x | ((unsigned)y << 16);
  }
  __syncthreads();
  size_t hb = (size_t)bs * kDpad;
  for (int d = t; d < kDpad; d += 256) {
    hx[hb + d] = (_Float16)(float)shx[d];
    hy[hb + d] = (_Float16)(float)shy[d];   // shy[240..319] == 0 (zero pad)
  }
}

// ---------------------------------------------------------------------------
// Kernel 2: T = hist(48xD) x Wt(Dx5) via V_WMMA_F32_16X16X32_F16 (exact: all
// operands are small integers), then 5x5 kernel over segment axis -> mean[s],
// aligned[s] = rint(mean - D/2), and median/MAD outlier flags.
// One workgroup per batch; 6 waves = {X,Y} coords x 3 M-tiles of 16 segments.
// The constant weight matrix Wt[c][col][K] is materialized once in LDS so the
// inner loop is pure loads + v_wmma (no per-element masking).
// ---------------------------------------------------------------------------
__global__ __launch_bounds__(192)
void k_mean(const _Float16* __restrict__ hx, const _Float16* __restrict__ hy,
            const float* __restrict__ kern,
            int* __restrict__ alX, int* __restrict__ alY,
            int* __restrict__ outl) {
  __shared__ alignas(16) _Float16 Wt[2][16][kWtPitch];
  __shared__ float T[2][kS][5];
  __shared__ float Mb[2][kS];
  int b = blockIdx.x;
  int t = threadIdx.x, wave = t >> 5, lane = t & 31;

  // Build the weight matrix once (branchless); only K<320 entries are read.
  for (int u = t; u < 2 * 16 * kDpad; u += 192) {
    int c2   = u / (16 * kDpad);
    int rem  = u - c2 * 16 * kDpad;
    int col2 = rem / kDpad;
    int K2   = rem - col2 * kDpad;
    int Dc2  = (c2 == 0) ? kW : kH;
    int wv   = K2 + 2 - col2;                // conv output coordinate
    bool ok  = (col2 < 5) & (K2 < Dc2) & (wv >= 0) & (wv < Dc2);
    Wt[c2][col2][K2] = ok ? (_Float16)(float)wv : (_Float16)0.f;
  }
  __syncthreads();

  int c = wave / 3, mtile = wave % 3;
  const _Float16* hist = ((c == 0) ? hx : hy) + (size_t)b * kS * kDpad;

  // A layout (16-bit A 16x32): lanes 0-15 -> M=row, K = {kb..kb+7, 16+kb..16+kb+7}
  // with kb=0 (lanes 0-15) / kb=8 (lanes 16-31).  B mirrored with N on lanes.
  int row = mtile * 16 + (lane & 15);
  int kb  = (lane < 16) ? 0 : 8;
  int col = lane & 15;
  const _Float16* rowp = hist + (size_t)row * kDpad;
  const _Float16* wrow = &Wt[c][col][0];

  v8f acc = {};
  for (int ks = 0; ks < 10; ++ks) {          // K = 320 = 10 * 32
    int Kb = ks * 32;
    v8h alo = *(const v8h*)(rowp + Kb + kb);
    v8h ahi = *(const v8h*)(rowp + Kb + 16 + kb);
    v8h blo = *(const v8h*)(wrow + Kb + kb);
    v8h bhi = *(const v8h*)(wrow + Kb + 16 + kb);
    v16h a, bm;
#pragma unroll
    for (int e = 0; e < 8; ++e) {
      a[e] = alo[e];  a[8 + e] = ahi[e];
      bm[e] = blo[e]; bm[8 + e] = bhi[e];
    }
    acc = __builtin_amdgcn_wmma_f32_16x16x32_f16(
        false, a, false, bm, (short)0, acc, false, false);
  }
  // C/D layout: VGPR r -> M = r (lanes 0-15) / 8+r (lanes 16-31), N = lane&15
  if (col < 5) {
    int mhi = (lane < 16) ? 0 : 8;
#pragma unroll
    for (int r = 0; r < 8; ++r) T[c][mtile * 16 + mhi + r][col] = acc[r];
  }
  __syncthreads();

  // mean[s] = sum_{i,j} k[i,j] * T[s+i-2, j] / segLen ; aligned = rint(mean - D/2)
  if (t < 96) {
    int cc = t / kS, s = t % kS;
    float m = 0.f;
    for (int i = 0; i < 5; ++i) {
      int r = s + i - 2;
      if (r >= 0 && r < kS)
        for (int j = 0; j < 5; ++j) m += kern[i * 5 + j] * T[cc][r][j];
    }
    m /= (float)kSegLen;
    Mb[cc][s] = m;
    float half = (cc == 0) ? 160.f : 120.f;
    int al = (int)rintf(m - half);           // round half-to-even, like jnp.round
    ((cc == 0) ? alX : alY)[b * kS + s] = al;
  }
  __syncthreads();

  // outlier flags: modified z-score of first element of each 10-window
  if (t < kNScan) {
    int flag = 0;
    for (int cc = 0; cc < 2; ++cc) {
      float win[10], sw[10], d[10];
      for (int k = 0; k < 10; ++k) { win[k] = Mb[cc][kSIDX + 1 + t + k]; sw[k] = win[k]; }
      for (int i = 1; i < 10; ++i) {         // insertion sort
        float v = sw[i]; int j = i - 1;
        while (j >= 0 && sw[j] > v) { sw[j + 1] = sw[j]; --j; }
        sw[j + 1] = v;
      }
      float med = 0.5f * (sw[4] + sw[5]);
      for (int k = 0; k < 10; ++k) d[k] = fabsf(win[k] - med);
      float d0 = d[0];
      for (int i = 1; i < 10; ++i) {
        float v = d[i]; int j = i - 1;
        while (j >= 0 && d[j] > v) { d[j + 1] = d[j]; --j; }
        d[j + 1] = v;
      }
      float mad = 0.5f * (d[4] + d[5]);
      float mz = 0.6745f * d0 / mad;         // inf/nan semantics match IEEE
      flag |= (mz > 3.0f) ? 1 : 0;
    }
    outl[b * kNScan + t] = flag;
  }
}

// ---------------------------------------------------------------------------
// Kernel 3: per-batch sequential scan (33 rows max, early stop), bitmask state
// in LDS, conf as u8 in LDS, container via global atomics (L2-resident),
// then mean+std(ddof=1) normalization of both images.
// One workgroup (1024 threads = 32 waves) per batch.
// ---------------------------------------------------------------------------
__global__ __launch_bounds__(1024)
void k_scan(const unsigned* __restrict__ packed,
            const int* __restrict__ alX, const int* __restrict__ alY,
            const int* __restrict__ outl,
            unsigned* __restrict__ container,
            float* __restrict__ out) {
  __shared__ unsigned vnewB[kMaskW], v0B[kMaskW], v1B[kMaskW];
  __shared__ unsigned char confB[kHW];
  __shared__ unsigned cntNew, cntTot;
  __shared__ double wred[4][32];
  __shared__ double fin[4];

  int b = blockIdx.x;
  int t = threadIdx.x;
  unsigned* cont = container + (size_t)b * kHW;
  const unsigned* pk = packed + (size_t)b * kNev;

  for (int d = t; d < kMaskW; d += 1024) { vnewB[d] = 0u; v0B[d] = 0u; v1B[d] = 0xFFFFFFFFu; }
  for (int p = t; p < kHW; p += 1024) confB[p] = 0;
  __syncthreads();

  // init with row SIDX: container hist + v0 mask, v1 = all ones
  {
    int ax = alX[b * kS + kSIDX], ay = alY[b * kS + kSIDX];
    for (int e = 0; e < 8; ++e) {
      unsigned v = pk[kSIDX * kSegLen + e * 1024 + t];
      int x = (int)(v & 0xFFFFu) - ax; x = x < 0 ? 0 : (x > kW - 1 ? kW - 1 : x);
      int y = (int)(v >> 16) - ay;     y = y < 0 ? 0 : (y > kH - 1 ? kH - 1 : y);
      int p = x + kW * y;
      atomicOr(&v0B[p >> 5], 1u << (p & 31));
      atomicAdd(&cont[p], 1u);
    }
  }
  __syncthreads();

  for (int s = kSIDX + 1; s < kEIDX; ++s) {
    if (outl[b * kNScan + (s - (kSIDX + 1))]) continue;   // uniform branch
    for (int d = t; d < kMaskW; d += 1024) vnewB[d] = 0u;
    if (t == 0) { cntNew = 0u; cntTot = 0u; }
    __syncthreads();
    int ax = alX[b * kS + s], ay = alY[b * kS + s];
    int px[8];
    for (int e = 0; e < 8; ++e) {
      unsigned v = pk[(size_t)s * kSegLen + e * 1024 + t];
      int x = (int)(v & 0xFFFFu) - ax; x = x < 0 ? 0 : (x > kW - 1 ? kW - 1 : x);
      int y = (int)(v >> 16) - ay;     y = y < 0 ? 0 : (y > kH - 1 ? kH - 1 : y);
      int p = x + kW * y;
      px[e] = p;
      atomicOr(&vnewB[p >> 5], 1u << (p & 31));
    }
    __syncthreads();
    unsigned nn = 0u, nt2 = 0u;
    for (int d = t; d < kMaskW; d += 1024) {
      unsigned nr = vnewB[d], o0 = v0B[d], o1 = v1B[d];
      nn  += __popc(nr & ~o0);                 // new_info bits
      nt2 += __popc(nr | o0);                  // |vnew|
      unsigned m = nr & o0 & o1;               // conf increments (owner thread)
      while (m) { int bp = __builtin_ctz(m); confB[d * 32 + bp]++; m &= m - 1u; }
    }
    if (nn)  atomicAdd(&cntNew, nn);
    if (nt2) atomicAdd(&cntTot, nt2);
    __syncthreads();
    float ratio = (float)cntNew / (float)cntTot;
    if (ratio < 0.01f) break;                  // "stopped" — uniform decision
    for (int e = 0; e < 8; ++e) atomicAdd(&cont[px[e]], 1u);
    for (int d = t; d < kMaskW; d += 1024) {
      unsigned o0 = v0B[d];
      v1B[d] = o0;
      v0B[d] = o0 | vnewB[d];
    }
    __syncthreads();
  }
  __syncthreads();
  __threadfence();   // make this block's global atomics visible to its own reads

  // normalize: clamp = mean + 3*std(ddof=1); out = min(v,clamp)/clamp
  double s1 = 0, s2 = 0, c1 = 0, c2 = 0;
  for (int p = t; p < kHW; p += 1024) {
    double v = (double)cont[p];
    double w = (double)confB[p];
    s1 += v; s2 += v * v; c1 += w; c2 += w * w;
  }
  int wave = t >> 5, lane = t & 31;
  for (int off = 16; off; off >>= 1) {         // wave32 shuffle reduction
    s1 += __shfl_down(s1, off, 32);
    s2 += __shfl_down(s2, off, 32);
    c1 += __shfl_down(c1, off, 32);
    c2 += __shfl_down(c2, off, 32);
  }
  if (lane == 0) { wred[0][wave] = s1; wred[1][wave] = s2; wred[2][wave] = c1; wred[3][wave] = c2; }
  __syncthreads();
  if (t < 4) { double a = 0; for (int w = 0; w < 32; ++w) a += wred[t][w]; fin[t] = a; }
  __syncthreads();
  double n = (double)kHW;
  double mC = fin[0] / n, vC = (fin[1] - n * mC * mC) / (n - 1.0);
  float clC = (float)(mC + 3.0 * sqrt(vC > 0 ? vC : 0));
  double mF = fin[2] / n, vF = (fin[3] - n * mF * mF) / (n - 1.0);
  float clF = (float)(mF + 3.0 * sqrt(vF > 0 ? vF : 0));
  float* outb = out + (size_t)b * 2 * kHW;     // (B, 2, H, W)
  for (int p = t; p < kHW; p += 1024) {
    float v = (float)cont[p];  v = v > clC ? clC : v;  outb[p]        = v / clC;
    float w = (float)confB[p]; w = w > clF ? clF : w;  outb[kHW + p]  = w / clF;
  }
}

// ---------------------------------------------------------------------------
extern "C" void kernel_launch(void* const* d_in, const int* in_sizes, int n_in,
                              void* d_out, int out_size, void* d_ws, size_t ws_size,
                              hipStream_t stream) {
  const int*   ev   = (const int*)d_in[0];     // (32, 393216, 5) int32
  const float* kern = (const float*)d_in[1];   // (5, 5) float32
  float* out = (float*)d_out;                  // (32, 2, 240, 320) float32

  char* ws = (char*)d_ws;
  size_t off = 0;
  unsigned* packed = (unsigned*)(ws + off);  off += (size_t)kB * kNev * 4;        // 50.3 MB
  _Float16* hx     = (_Float16*)(ws + off); off += (size_t)kB * kS * kDpad * 2;   // 983 KB
  _Float16* hy     = (_Float16*)(ws + off); off += (size_t)kB * kS * kDpad * 2;   // 983 KB
  int* alX  = (int*)(ws + off); off += (size_t)kB * kS * 4;
  int* alY  = (int*)(ws + off); off += (size_t)kB * kS * 4;
  int* outl = (int*)(ws + off); off += (size_t)kB * kNScan * 4;
  off = (off + 255) & ~(size_t)255;
  unsigned* container = (unsigned*)(ws + off); off += (size_t)kB * kHW * 4;       // 9.8 MB

  size_t nz = (size_t)kB * kHW;
  k_zero<<<(unsigned)((nz + 255) / 256), 256, 0, stream>>>(container, nz);
  k_hist<<<kB * kS, 256, 0, stream>>>(ev, packed, hx, hy);
  k_mean<<<kB, 192, 0, stream>>>(hx, hy, kern, alX, alY, outl);
  k_scan<<<kB, 1024, 0, stream>>>(packed, alX, alY, outl, container, out);
}
